// CRF_28793460752591
// MI455X (gfx1250) — matrix operand order repaired and checked by reading, not compiled, not run on hardware
//
#include <hip/hip_runtime.h>
#include <hip/hip_bf16.h>

// CRF forward-score kernel for MI455X (gfx1250, wave32, WMMA).
//
// Recurrence rewritten as exp-domain GEMM:
//   alpha'[b,i] = feat[b,t,i] + m_b + log( sum_j exp(T[i,j]) * exp(alpha[b,j]-m_b) )
// -> one 128x128 (f16) x 128x16 (f16) -> 128x16 (f32) matmul per workgroup
//    per time step, done with V_WMMA_F32_16X16X32_F16.

typedef __attribute__((ext_vector_type(16))) _Float16 v16h;
typedef __attribute__((ext_vector_type(8)))  float    v8f;

#define CRF_B 256
#define CRF_L 1024
#define CRF_C 128
#define WG_B  16      // batches per workgroup
#define NTHR  256     // 8 waves of 32

__global__ __launch_bounds__(NTHR)
void crf_forward_kernel(const float* __restrict__ feats,
                        const float* __restrict__ T,
                        float* __restrict__ fs /* [CRF_B] forward scores */) {
    // LDS: exp(T) staged once; alpha [j][b]; p in WMMA-B-fragment order; reductions.
    __shared__ __align__(32) _Float16 Eh[CRF_C * CRF_C];     // 32 KB
    __shared__ float  alpha[CRF_C * WG_B];                    // 8 KB, [j*16+b]
    __shared__ __align__(32) _Float16 pbuf[4 * 32 * WG_B];    // 4 KB, fragment order
    __shared__ float  pmax[16 * WG_B];                        // partial maxima [jg*16+b]
    __shared__ float  psum[16 * WG_B];

    const int tid  = threadIdx.x;
    const int lane = tid & 31;
    const int wave = tid >> 5;              // 0..7 -> i-rows [16*wave, 16*wave+16)
    const int b0   = blockIdx.x * WG_B;

    // ---- stage exp(T) into LDS as f16 (64 values/thread) ----
    for (int k = tid; k < CRF_C * CRF_C; k += NTHR)
        Eh[k] = (_Float16)__expf(T[k]);

    // ---- alpha init: alpha[j][b] = feats[b0+b, 0, j] ----
    for (int k = tid; k < CRF_C * WG_B; k += NTHR) {
        int b = k >> 7, j = k & 127;
        alpha[j * WG_B + b] = feats[(size_t)(b0 + b) * CRF_L * CRF_C + j];
    }
    __syncthreads();

    // ---- load per-wave A fragments of exp(T): rows M = 16*wave + (lane&15) ----
    // A in contiguous k order (same as B): lanes 0-15 hold K = 0..15, lanes 16-31 hold K = 16..31.
    v16h afrag[4];
    {
        const int Mrow = wave * 16 + (lane & 15);
        const int khi  = (lane >> 4) << 4;
        for (int kk = 0; kk < 4; ++kk)
            for (int e = 0; e < 16; ++e) {
                int kl = e + khi;     // 0..31
                afrag[kk][e] = Eh[Mrow * CRF_C + kk * 32 + kl];
            }
    }

    const int bcol  = tid & 15;                  // this thread's batch column
    const int jg    = tid >> 4;                  // j-group for partial max (8 j's)
    const int ibase = wave * 16 + ((lane >> 4) << 3);   // first of 8 output rows
    // NOTE: in the WMMA D layout this lane's column is lane&15 == tid&15 == bcol,
    // so the locally-reduced m below is exactly the m needed for the alpha update.

    for (int t = 1; t < CRF_L; ++t) {
        // ---- partial max over this thread's 8 j's for column bcol ----
        {
            float mx = -3.0e38f;
            #pragma unroll
            for (int j = jg * 8; j < jg * 8 + 8; ++j)
                mx = fmaxf(mx, alpha[j * WG_B + bcol]);
            pmax[jg * WG_B + bcol] = mx;
        }
        __syncthreads();                                    // B1

        // ---- every thread reduces the 16 partials for its own column ----
        float m = pmax[bcol];
        #pragma unroll
        for (int g = 1; g < 16; ++g) m = fmaxf(m, pmax[g * WG_B + bcol]);

        // ---- p[j][bcol] = exp(alpha - m), written in B-fragment order:
        //      pbuf[kk*512 + laneB*16 + e] <-> B[k = 32*kk + 16*(laneB>=16) + e][n = laneB&15]
        #pragma unroll
        for (int it = 0; it < (CRF_C * WG_B) / NTHR; ++it) {
            int j = (tid >> 4) + it * 16;
            float pv = __expf(alpha[j * WG_B + bcol] - m);
            int kk = j >> 5, r = j & 31;
            int laneB = ((r >> 4) << 4) | bcol;
            int e = r & 15;
            pbuf[kk * 512 + laneB * 16 + e] = (_Float16)pv;
        }
        __syncthreads();                                    // B2

        // ---- load all 4 B fragments first, then 4 back-to-back WMMAs ----
        v16h bfrag[4];
        #pragma unroll
        for (int kk = 0; kk < 4; ++kk)
            bfrag[kk] = *(const v16h*)&pbuf[kk * 512 + lane * 16];
        v8f acc = {};
        #pragma unroll
        for (int kk = 0; kk < 4; ++kk)
            acc = __builtin_amdgcn_wmma_f32_16x16x32_f16(
                false, afrag[kk], false, bfrag[kk], (short)0, acc, false, false);

        // ---- new alpha: D layout -> lane holds rows ibase..ibase+7, col bcol ----
        const float* frow = feats + (size_t)(b0 + bcol) * CRF_L * CRF_C
                                  + (size_t)t * CRF_C + ibase;
        if (t + 1 < CRF_L) __builtin_prefetch(frow + CRF_C, 0, 3);
        float4 f0 = *(const float4*)(frow);
        float4 f1 = *(const float4*)(frow + 4);
        float fv[8] = {f0.x, f0.y, f0.z, f0.w, f1.x, f1.y, f1.z, f1.w};
        #pragma unroll
        for (int v = 0; v < 8; ++v)
            alpha[(ibase + v) * WG_B + bcol] = fv[v] + m + __logf(acc[v]);
        __syncthreads();                                    // B3
    }

    // ---- forward_score[b] = logsumexp_j(alpha[j][b]), deterministic ----
    {
        float mx = -3.0e38f;
        #pragma unroll
        for (int j = jg * 8; j < jg * 8 + 8; ++j)
            mx = fmaxf(mx, alpha[j * WG_B + bcol]);
        pmax[jg * WG_B + bcol] = mx;
    }
    __syncthreads();
    {
        float m = pmax[bcol];
        #pragma unroll
        for (int g = 1; g < 16; ++g) m = fmaxf(m, pmax[g * WG_B + bcol]);
        float s = 0.f;
        #pragma unroll
        for (int j = jg * 8; j < jg * 8 + 8; ++j)
            s += __expf(alpha[j * WG_B + bcol] - m);
        psum[jg * WG_B + bcol] = s;
        __syncthreads();
        if (tid < WG_B) {
            float ss = psum[tid];
            #pragma unroll
            for (int g = 1; g < 16; ++g) ss += psum[g * WG_B + tid];
            fs[b0 + tid] = m + __logf(ss);    // tid<16 => bcol==tid, m is column tid's max
        }
    }
}

// gold[b] = sum_t feats[b,t,tags[b,t]] + sum_{t>=1} T[tags[b,t], tags[b,t-1]]
__global__ __launch_bounds__(NTHR)
void crf_gold_kernel(const float* __restrict__ feats,
                     const int* __restrict__ tags,
                     const float* __restrict__ T,
                     float* __restrict__ gold) {
    __shared__ float red[NTHR];
    const int b = blockIdx.x, tid = threadIdx.x;
    const float* fb = feats + (size_t)b * CRF_L * CRF_C;
    const int*   tb = tags  + (size_t)b * CRF_L;
    float s = 0.f;
    for (int t = tid; t < CRF_L; t += NTHR) {
        int tg = tb[t];
        s += fb[(size_t)t * CRF_C + tg];
        if (t >= 1) s += T[(size_t)tg * CRF_C + tb[t - 1]];
    }
    red[tid] = s;
    __syncthreads();
    for (int off = NTHR / 2; off > 0; off >>= 1) {
        if (tid < off) red[tid] += red[tid + off];
        __syncthreads();
    }
    if (tid == 0) gold[b] = red[0];
}

// out = mean_b(fs[b] - gold[b]); B == 256 == blockDim
__global__ __launch_bounds__(NTHR)
void crf_final_kernel(const float* __restrict__ fs,
                      const float* __restrict__ gold,
                      float* __restrict__ out) {
    __shared__ float red[NTHR];
    const int tid = threadIdx.x;
    red[tid] = fs[tid] - gold[tid];
    __syncthreads();
    for (int off = NTHR / 2; off > 0; off >>= 1) {
        if (tid < off) red[tid] += red[tid + off];
        __syncthreads();
    }
    if (tid == 0) out[0] = red[0] * (1.0f / CRF_B);
}

extern "C" void kernel_launch(void* const* d_in, const int* in_sizes, int n_in,
                              void* d_out, int out_size, void* d_ws, size_t ws_size,
                              hipStream_t stream) {
    const float* feats = (const float*)d_in[0];   // (256,1024,128) f32
    const int*   tags  = (const int*)  d_in[1];   // (256,1024) i32
    const float* T     = (const float*)d_in[2];   // (128,128) f32
    float* out  = (float*)d_out;                  // scalar f32
    float* fs   = (float*)d_ws;                   // [256]
    float* gold = fs + CRF_B;                     // [256]

    crf_forward_kernel<<<CRF_B / WG_B, NTHR, 0, stream>>>(feats, T, fs);
    crf_gold_kernel<<<CRF_B, NTHR, 0, stream>>>(feats, tags, T, gold);
    crf_final_kernel<<<1, NTHR, 0, stream>>>(fs, gold, out);
}
